// PONITA_63797444215147
// MI455X (gfx1250) — matrix operand order repaired and checked by reading, not compiled, not run on hardware
//
#include <hip/hip_runtime.h>

// ---------------- problem constants ----------------
#define N_NODES 8000
#define N_EDGES 64000
#define NB      64
#define NGRID   12
#define HID     128
#define WIDEH   512
#define NL      3

#define AT_STRIDE 136   // 16x128 bf16 LDS tile row stride (halves), padded
#define TT_STRIDE 520   // 16x512 bf16 LDS tile row stride (halves), padded

typedef __attribute__((ext_vector_type(16))) __bf16       v16bf;
typedef __attribute__((ext_vector_type(8)))  float        v8f;
typedef __attribute__((ext_vector_type(8)))  unsigned int v8u;

#define WMMA_BF16(a,b,c) \
  __builtin_amdgcn_wmma_f32_16x16x32_bf16(false,(a),false,(b),(short)0,(c),false,false)

// ---------------- helpers ----------------
__device__ __forceinline__ unsigned short f2bf(float f) {
  unsigned int u = __float_as_uint(f);
  u += 0x7FFFu + ((u >> 16) & 1u);          // round-to-nearest-even
  return (unsigned short)(u >> 16);
}

__device__ __forceinline__ float gelu_f(float x) {
  return 0.5f * x * (1.0f + erff(x * 0.7071067811865476f));
}

// wave-local LDS RAW/WAR fence: all DS ops of this wave complete
__device__ __forceinline__ void wave_sync_lds() {
  __builtin_amdgcn_wave_barrier();
  asm volatile("s_wait_dscnt 0" ::: "memory");
  __builtin_amdgcn_wave_barrier();
}

// A fragment (16xK bf16, ISA 7.12.2 layout) gathered from an LDS tile
__device__ __forceinline__ v16bf load_a_frag(const unsigned short* lds, int stride,
                                             int lane, int kbase) {
  int m  = lane & 15;
  int hi = lane >> 4;
  v8u u;
#pragma unroll
  for (int v = 0; v < 8; ++v) {
    int k = kbase + ((v < 4) ? (hi * 8 + 2 * v) : (16 + hi * 8 + 2 * (v - 4)));
    u[v] = *(const unsigned int*)(lds + m * stride + k);
  }
  return __builtin_bit_cast(v16bf, u);
}

// B fragment from pre-swizzled global weights: lane-contiguous 32B slices
__device__ __forceinline__ v16bf load_b_frag(const unsigned short* g, int fragIdx, int lane) {
  const v8u* p = (const v8u*)(g + (size_t)fragIdx * 512 + lane * 16);
  v8u u = *p;
  return __builtin_bit_cast(v16bf, u);
}

__device__ __forceinline__ void grid0_row(int n, float g[3]) {
  const float golden = 1.6180339887498949f;
  float th = 6.283185307179586f * (float)n / golden;
  float z  = 1.f - (2.f * (float)n + 1.f) / (float)NGRID;
  float r  = sqrtf(fmaxf(1.f - z * z, 0.f));
  g[0] = r * cosf(th); g[1] = r * sinf(th); g[2] = z;
}

// ---------------- utility kernels ----------------
__global__ void k_zero(float* __restrict__ p, int n) {
  int i = blockIdx.x * blockDim.x + threadIdx.x;
  int stride = gridDim.x * blockDim.x;
  for (; i < n; i += stride) p[i] = 0.f;
}

// convert row-major f32 KxN weight -> bf16 WMMA-B fragment layout
// frag(kc,nt): lane holds B[kc*32 + (h<8 ? (lane>>4)*8 + h : 16 + (lane>>4)*8 + h-8)][nt*16 + (lane&15)], h=0..15
__global__ void k_convert(const float* __restrict__ src, unsigned short* __restrict__ dst,
                          int K, int Keff, int N) {
  int t = blockIdx.x * blockDim.x + threadIdx.x;
  int total = K * N;
  if (t >= total) return;
  int h       = t & 15;
  int lane    = (t >> 4) & 31;
  int fragIdx = t >> 9;
  int Nt = N >> 4;
  int nt = fragIdx % Nt;
  int kc = fragIdx / Nt;
  int k = kc * 32 + ((h < 8) ? ((lane >> 4) * 8 + h) : (16 + (lane >> 4) * 8 + (h - 8)));
  int n = nt * 16 + (lane & 15);
  float val = (k < Keff) ? src[k * N + n] : 0.f;
  dst[t] = f2bf(val);
}

// node_grid[node,n,:] = Q[batch[node]] @ grid0[n]
__global__ void k_prep(const float* __restrict__ Q, const int* __restrict__ batch,
                       float* __restrict__ ng) {
  int t = blockIdx.x * blockDim.x + threadIdx.x;
  if (t >= N_NODES * NGRID) return;
  int node = t / NGRID, n = t - node * NGRID;
  float g[3]; grid0_row(n, g);
  const float* q = Q + batch[node] * 9;
  ng[t * 3 + 0] = q[0] * g[0] + q[1] * g[1] + q[2] * g[2];
  ng[t * 3 + 1] = q[3] * g[0] + q[4] * g[1] + q[5] * g[2];
  ng[t * 3 + 2] = q[6] * g[0] + q[7] * g[1] + q[8] * g[2];
}

// h0 = [x | vec . node_grid] @ Wemb   (18 -> 128)
__global__ __launch_bounds__(128) void k_embed(const float* __restrict__ x,
                                               const float* __restrict__ vec,
                                               const float* __restrict__ ng,
                                               const float* __restrict__ Wemb,
                                               float* __restrict__ h) {
  int r = blockIdx.x;                    // node*NGRID + n
  int node = r / NGRID;
  int c = threadIdx.x;
  const float* g = ng + r * 3;
  const float* xn = x + node * 16;
  float s = 0.f;
#pragma unroll
  for (int i = 0; i < 16; ++i) s += xn[i] * Wemb[i * HID + c];
  const float* v = vec + node * 6;
  float xv0 = v[0] * g[0] + v[1] * g[1] + v[2] * g[2];
  float xv1 = v[3] * g[0] + v[4] * g[1] + v[5] * g[2];
  s += xv0 * Wemb[16 * HID + c] + xv1 * Wemb[17 * HID + c];
  h[r * HID + c] = s;
}

// kb_sh[p,o,:] = gelu(gelu(poly3(grid0[p].grid0[o]) @ Wsh1 + b) @ Wsh2 + b)
__global__ __launch_bounds__(128) void k_sh(const float* __restrict__ Wsh1,
                                            const float* __restrict__ bsh1,
                                            const float* __restrict__ Wsh2,
                                            const float* __restrict__ bsh2,
                                            float* __restrict__ kb_sh) {
  __shared__ float t1s[HID];
  int blk = blockIdx.x;                  // p*12 + o
  int p = blk / NGRID, o = blk - p * NGRID;
  int c = threadIdx.x;
  float gp[3], go[3];
  grid0_row(p, gp); grid0_row(o, go);
  float a = gp[0] * go[0] + gp[1] * go[1] + gp[2] * go[2];
  float f0 = a, f1 = a * a, f2 = a * a * a;
  float t1 = gelu_f(f0 * Wsh1[c] + f1 * Wsh1[HID + c] + f2 * Wsh1[2 * HID + c] + bsh1[c]);
  t1s[c] = t1;
  __syncthreads();
  float s = bsh2[c];
  for (int k = 0; k < HID; ++k) s += t1s[k] * Wsh2[k * HID + c];
  kb_sh[blk * HID + c] = gelu_f(s);
}

// fk[l,p,o,:] = kb_sh[p,o,:] @ Wfk[l]
__global__ __launch_bounds__(128) void k_fk(const float* __restrict__ kb_sh,
                                            const float* __restrict__ Wfk,
                                            float* __restrict__ fk) {
  __shared__ float row[HID];
  int blk = blockIdx.x;                  // l*144 + r
  int l = blk / 144, r = blk - l * 144;
  int c = threadIdx.x;
  row[c] = kb_sh[r * HID + c];
  __syncthreads();
  float s = 0.f;
  for (int k = 0; k < HID; ++k) s += row[k] * Wfk[(l * HID + k) * HID + c];
  fk[blk * HID + c] = s;
}

// ---------------- fused edge kernel (per layer) ----------------
// rows = (edge, grid) pairs; 16-row tile per wave; 4 waves per WG
// attr -> poly(14) -> x Wsp1 -> gelu -> x Wsp2 -> gelu -> x Wk[l]
// msg = h[send]*k ; atomicAdd agg[recv]   (h/agg are L2-resident)
__global__ __launch_bounds__(128) void k_edge(
    const float* __restrict__ pos, const int* __restrict__ ei,
    const float* __restrict__ ng, const float* __restrict__ h,
    const unsigned short* __restrict__ wsp1f, const float* __restrict__ bsp1,
    const unsigned short* __restrict__ wsp2f, const float* __restrict__ bsp2,
    const unsigned short* __restrict__ wkf,
    float* __restrict__ agg) {
  __shared__ unsigned short aT[4][16 * AT_STRIDE];
  __shared__ int sIdx[4][16];
  __shared__ int rIdx[4][16];
  int w = threadIdx.x >> 5;
  int lane = threadIdx.x & 31;
  int rowBase = (blockIdx.x * 4 + w) * 16;
  unsigned short* a = aT[w];
  int col = lane & 15, hi = lane >> 4;

  if (lane < 16) {
    int row = rowBase + lane;
    int e = row / NGRID;
    int n = row - e * NGRID;
    int s = ei[e];
    int r = ei[N_EDGES + e];
    sIdx[w][lane] = s;
    rIdx[w][lane] = r;
    float rx = pos[3 * s + 0] - pos[3 * r + 0];
    float ry = pos[3 * s + 1] - pos[3 * r + 1];
    float rz = pos[3 * s + 2] - pos[3 * r + 2];
    const float* g = ng + (r * NGRID + n) * 3;
    float gx = g[0], gy = g[1], gz = g[2];
    float i1 = rx * gx + ry * gy + rz * gz;
    float dx = rx - i1 * gx, dy = ry - i1 * gy, dz = rz - i1 * gz;
    float i2 = sqrtf(dx * dx + dy * dy + dz * dz);
    float ft[14];
    ft[0] = i1; ft[1] = i2;
    ft[2] = i1 * i1; ft[3] = i1 * i2; ft[4] = i2 * i1; ft[5] = i2 * i2;
    ft[6]  = ft[2] * i1; ft[7]  = ft[2] * i2;
    ft[8]  = ft[3] * i1; ft[9]  = ft[3] * i2;
    ft[10] = ft[4] * i1; ft[11] = ft[4] * i2;
    ft[12] = ft[5] * i1; ft[13] = ft[5] * i2;
#pragma unroll
    for (int k = 0; k < 14; ++k) a[lane * AT_STRIDE + k] = f2bf(ft[k]);
#pragma unroll
    for (int k = 14; k < 32; ++k) a[lane * AT_STRIDE + k] = 0;
  }
  wave_sync_lds();

  v16bf a0 = load_a_frag(a, AT_STRIDE, lane, 0);
  wave_sync_lds();

  // basis MLP layer 1: poly(pad32) x Wsp1(32x128)
#pragma unroll
  for (int nt = 0; nt < 8; ++nt) {
    v8f c = {};
    c = WMMA_BF16(a0, load_b_frag(wsp1f, nt, lane), c);
    float bias = bsp1[nt * 16 + col];
#pragma unroll
    for (int v = 0; v < 8; ++v) {
      int m = v + 8 * hi;
      a[m * AT_STRIDE + nt * 16 + col] = f2bf(gelu_f(c[v] + bias));
    }
  }
  wave_sync_lds();

  v16bf a1[4];
#pragma unroll
  for (int kc = 0; kc < 4; ++kc) a1[kc] = load_a_frag(a, AT_STRIDE, lane, kc * 32);
  wave_sync_lds();

  // basis MLP layer 2: T1(16x128) x Wsp2(128x128)
#pragma unroll
  for (int nt = 0; nt < 8; ++nt) {
    v8f c = {};
#pragma unroll
    for (int kc = 0; kc < 4; ++kc)
      c = WMMA_BF16(a1[kc], load_b_frag(wsp2f, kc * 8 + nt, lane), c);
    float bias = bsp2[nt * 16 + col];
#pragma unroll
    for (int v = 0; v < 8; ++v) {
      int m = v + 8 * hi;
      a[m * AT_STRIDE + nt * 16 + col] = f2bf(gelu_f(c[v] + bias));
    }
  }
  wave_sync_lds();

  v16bf a2[4];
#pragma unroll
  for (int kc = 0; kc < 4; ++kc) a2[kc] = load_a_frag(a, AT_STRIDE, lane, kc * 32);
  wave_sync_lds();

  // k = kb_sp x Wk[l]; msg = h[send]*k; scatter-add agg[recv]
#pragma unroll
  for (int nt = 0; nt < 8; ++nt) {
    v8f c = {};
#pragma unroll
    for (int kc = 0; kc < 4; ++kc)
      c = WMMA_BF16(a2[kc], load_b_frag(wkf, kc * 8 + nt, lane), c);
    int ch = nt * 16 + col;
#pragma unroll
    for (int v = 0; v < 8; ++v) {
      int m = v + 8 * hi;
      int row = rowBase + m;
      int n = row - (row / NGRID) * NGRID;
      int s = sIdx[w][m];
      int r = rIdx[w][m];
      float msg = h[(s * NGRID + n) * HID + ch] * c[v];
      atomicAdd(&agg[(r * NGRID + n) * HID + ch], msg);
    }
  }
}

// ---------------- fused node kernel (per layer) ----------------
// sphere conv (12x12 per channel) + LayerNorm (VALU) + MLP 128->512->128 (WMMA)
// + residual + readout accumulation
__global__ __launch_bounds__(128) void k_node(
    const float* __restrict__ agg, const float* __restrict__ fk,
    const float* __restrict__ bconv, const float* __restrict__ lng,
    const float* __restrict__ lnb,
    const unsigned short* __restrict__ w1f, const float* __restrict__ b1,
    const unsigned short* __restrict__ w2f, const float* __restrict__ b2,
    const float* __restrict__ Wro, const float* __restrict__ bro,
    float* __restrict__ h, float* __restrict__ ro) {
  __shared__ float fbuf[NGRID * HID];               // x2, later h_out
  __shared__ unsigned short hnT[16 * AT_STRIDE];
  __shared__ unsigned short tT[16 * TT_STRIDE];
  __shared__ float mu[NGRID], rs[NGRID];
  int b = blockIdx.x;
  int tid = threadIdx.x;
  int c = tid;

  float av[NGRID];
#pragma unroll
  for (int o = 0; o < NGRID; ++o) av[o] = agg[(b * NGRID + o) * HID + c];
  float bc = bconv[c];
#pragma unroll
  for (int p = 0; p < NGRID; ++p) {
    float s = 0.f;
#pragma unroll
    for (int o = 0; o < NGRID; ++o) s += av[o] * fk[(p * NGRID + o) * HID + c];
    fbuf[p * HID + c] = s * (1.f / NGRID) + bc;
  }
  __syncthreads();
  if (tid < NGRID) {
    float m = 0.f;
    for (int k = 0; k < HID; ++k) m += fbuf[tid * HID + k];
    m *= (1.f / HID);
    float v = 0.f;
    for (int k = 0; k < HID; ++k) { float d = fbuf[tid * HID + k] - m; v += d * d; }
    v *= (1.f / HID);
    mu[tid] = m; rs[tid] = rsqrtf(v + 1e-5f);
  }
  __syncthreads();
  float g = lng[c], bb = lnb[c];
#pragma unroll
  for (int p = 0; p < NGRID; ++p)
    hnT[p * AT_STRIDE + c] = f2bf((fbuf[p * HID + c] - mu[p]) * rs[p] * g + bb);
#pragma unroll
  for (int p = NGRID; p < 16; ++p) hnT[p * AT_STRIDE + c] = 0;
  __syncthreads();

  int w = tid >> 5, lane = tid & 31, col = lane & 15, hi = lane >> 4;
  v16bf a1[4];
#pragma unroll
  for (int kc = 0; kc < 4; ++kc) a1[kc] = load_a_frag(hnT, AT_STRIDE, lane, kc * 32);

  // MLP layer 1: 16x128 x 128x512 (8 N-tiles per wave)
#pragma unroll
  for (int i = 0; i < 8; ++i) {
    int nt = w * 8 + i;
    v8f cc = {};
#pragma unroll
    for (int kc = 0; kc < 4; ++kc)
      cc = WMMA_BF16(a1[kc], load_b_frag(w1f, kc * 32 + nt, lane), cc);
    float bias = b1[nt * 16 + col];
#pragma unroll
    for (int v = 0; v < 8; ++v) {
      int m = v + 8 * hi;
      tT[m * TT_STRIDE + nt * 16 + col] = f2bf(gelu_f(cc[v] + bias));
    }
  }
  __syncthreads();

  // MLP layer 2: 16x512 x 512x128 (2 N-tiles per wave) + residual
#pragma unroll
  for (int i = 0; i < 2; ++i) {
    int nt = w * 2 + i;
    v8f cc = {};
#pragma unroll
    for (int kc = 0; kc < 16; ++kc) {
      v16bf a2 = load_a_frag(tT, TT_STRIDE, lane, kc * 32);
      cc = WMMA_BF16(a2, load_b_frag(w2f, kc * 8 + nt, lane), cc);
    }
    int ch = nt * 16 + col;
    float bias = b2[ch];
#pragma unroll
    for (int v = 0; v < 8; ++v) {
      int m = v + 8 * hi;
      if (m < NGRID) {
        float hout = h[(b * NGRID + m) * HID + ch] + cc[v] + bias;
        h[(b * NGRID + m) * HID + ch] = hout;
        fbuf[m * HID + ch] = hout;
      }
    }
  }
  __syncthreads();

  // readout: ro += (h_out @ Wro + bro)/L   (128 -> 9)
  if (tid < NGRID * 9) {
    int n = tid / 9, j = tid - n * 9;
    float s = 0.f;
    for (int k = 0; k < HID; ++k) s += fbuf[n * HID + k] * Wro[k * 9 + j];
    ro[(b * NGRID + n) * 9 + j] += (s + bro[j]) * (1.f / NL);
  }
}

// ---------------- final readout ----------------
__global__ __launch_bounds__(32) void k_final(const float* __restrict__ ro,
                                              const float* __restrict__ ng,
                                              const int* __restrict__ batch,
                                              float* __restrict__ out) {
  int node = blockIdx.x;
  int t = threadIdx.x;
  int b = batch[node];
  if (t < 8) {
    float s = 0.f;
#pragma unroll
    for (int n = 0; n < NGRID; ++n) s += ro[(node * NGRID + n) * 9 + t];
    atomicAdd(&out[b * 8 + t], s * (1.f / NGRID));
  } else if (t < 11) {
    int d = t - 8;
    float s = 0.f;
#pragma unroll
    for (int n = 0; n < NGRID; ++n)
      s += ro[(node * NGRID + n) * 9 + 8] * ng[(node * NGRID + n) * 3 + d];
    atomicAdd(&out[NB * 8 + b * 3 + d], s * (1.f / NGRID));
  }
}

// ---------------- host orchestration ----------------
extern "C" void kernel_launch(void* const* d_in, const int* in_sizes, int n_in,
                              void* d_out, int out_size, void* d_ws, size_t ws_size,
                              hipStream_t stream) {
  (void)in_sizes; (void)n_in; (void)ws_size;
  const float* x    = (const float*)d_in[0];
  const float* vec  = (const float*)d_in[1];
  const float* pos  = (const float*)d_in[2];
  const float* Q    = (const float*)d_in[3];
  const float* Wsp1 = (const float*)d_in[4];
  const float* bsp1 = (const float*)d_in[5];
  const float* Wsp2 = (const float*)d_in[6];
  const float* bsp2 = (const float*)d_in[7];
  const float* Wsh1 = (const float*)d_in[8];
  const float* bsh1 = (const float*)d_in[9];
  const float* Wsh2 = (const float*)d_in[10];
  const float* bsh2 = (const float*)d_in[11];
  const float* Wemb = (const float*)d_in[12];
  const float* Wk   = (const float*)d_in[13];
  const float* Wfk  = (const float*)d_in[14];
  const float* bconv= (const float*)d_in[15];
  const float* ln_g = (const float*)d_in[16];
  const float* ln_b = (const float*)d_in[17];
  const float* W1   = (const float*)d_in[18];
  const float* b1   = (const float*)d_in[19];
  const float* W2   = (const float*)d_in[20];
  const float* b2   = (const float*)d_in[21];
  const float* Wro  = (const float*)d_in[22];
  const float* bro  = (const float*)d_in[23];
  const int* ei     = (const int*)d_in[24];
  const int* batch  = (const int*)d_in[25];
  float* out = (float*)d_out;

  // workspace carve-up (~104 MB total)
  char* ws = (char*)d_ws;
  size_t off = 0;
  auto alloc = [&](size_t bytes) { size_t o = off; off = (off + bytes + 255) & ~(size_t)255; return o; };
  float* ng     = (float*)(ws + alloc((size_t)N_NODES * NGRID * 3 * 4));
  float* h      = (float*)(ws + alloc((size_t)N_NODES * NGRID * HID * 4));
  float* agg    = (float*)(ws + alloc((size_t)N_NODES * NGRID * HID * 4));
  float* ro     = (float*)(ws + alloc((size_t)N_NODES * NGRID * 9 * 4));
  float* kb_sh  = (float*)(ws + alloc((size_t)144 * HID * 4));
  float* fk     = (float*)(ws + alloc((size_t)NL * 144 * HID * 4));
  unsigned short* wsp1f = (unsigned short*)(ws + alloc((size_t)32 * HID * 2));
  unsigned short* wsp2f = (unsigned short*)(ws + alloc((size_t)HID * HID * 2));
  unsigned short* wkf   = (unsigned short*)(ws + alloc((size_t)NL * HID * HID * 2));
  unsigned short* w1f   = (unsigned short*)(ws + alloc((size_t)NL * HID * WIDEH * 2));
  unsigned short* w2f   = (unsigned short*)(ws + alloc((size_t)NL * WIDEH * HID * 2));

  // zero accumulators / output
  k_zero<<<16, 256, 0, stream>>>(out, out_size);
  k_zero<<<1024, 256, 0, stream>>>(ro, N_NODES * NGRID * 9);

  // bf16 weight conversion + WMMA fragment pre-swizzle
  k_convert<<<(32 * HID + 255) / 256, 256, 0, stream>>>(Wsp1, wsp1f, 32, 14, HID);
  k_convert<<<(HID * HID + 255) / 256, 256, 0, stream>>>(Wsp2, wsp2f, HID, HID, HID);
  for (int l = 0; l < NL; ++l) {
    k_convert<<<(HID * HID + 255) / 256, 256, 0, stream>>>(
        Wk + (size_t)l * HID * HID, wkf + (size_t)l * HID * HID, HID, HID, HID);
    k_convert<<<(HID * WIDEH + 255) / 256, 256, 0, stream>>>(
        W1 + (size_t)l * HID * WIDEH, w1f + (size_t)l * HID * WIDEH, HID, HID, WIDEH);
    k_convert<<<(WIDEH * HID + 255) / 256, 256, 0, stream>>>(
        W2 + (size_t)l * WIDEH * HID, w2f + (size_t)l * WIDEH * HID, WIDEH, WIDEH, HID);
  }

  // geometry + embedding + spherical basis
  k_prep<<<(N_NODES * NGRID + 255) / 256, 256, 0, stream>>>(Q, batch, ng);
  k_embed<<<N_NODES * NGRID, 128, 0, stream>>>(x, vec, ng, Wemb, h);
  k_sh<<<144, 128, 0, stream>>>(Wsh1, bsh1, Wsh2, bsh2, kb_sh);
  k_fk<<<NL * 144, 128, 0, stream>>>(kb_sh, Wfk, fk);

  // layers
  const int edgeTiles = (N_EDGES * NGRID) / 16;   // 48000
  for (int l = 0; l < NL; ++l) {
    k_zero<<<2048, 256, 0, stream>>>(agg, N_NODES * NGRID * HID);
    k_edge<<<edgeTiles / 4, 128, 0, stream>>>(
        pos, ei, ng, h, wsp1f, bsp1, wsp2f, bsp2,
        wkf + (size_t)l * HID * HID, agg);
    k_node<<<N_NODES, 128, 0, stream>>>(
        agg, fk + (size_t)l * 144 * HID, bconv + l * HID,
        ln_g + l * HID, ln_b + l * HID,
        w1f + (size_t)l * HID * WIDEH, b1 + l * WIDEH,
        w2f + (size_t)l * WIDEH * HID, b2 + l * HID,
        Wro + (size_t)l * HID * 9, bro + l * 9,
        h, ro);
  }

  k_final<<<N_NODES, 32, 0, stream>>>(ro, ng, batch, out);
}
